// SampleAndPredict_2808908612291
// MI455X (gfx1250) — matrix operand order repaired and checked by reading, not compiled, not run on hardware
//
#include <hip/hip_runtime.h>

typedef __attribute__((ext_vector_type(16))) _Float16 f16x16;
typedef __attribute__((ext_vector_type(8)))  _Float16 f16x8;
typedef __attribute__((ext_vector_type(4)))  _Float16 f16x4;
typedef __attribute__((ext_vector_type(8)))  float    f32x8;
typedef __attribute__((ext_vector_type(4)))  float    f32x4;

#define S_  2048
#define N_  8
#define E_  512
#define C_  1024
#define G_  32
#define CG_ 32
#define M_  (S_ * N_)   // 16384 rows

// ---------------------------------------------------------------------------
// Fragment loaders (wave32; contiguous k order used for both A and B, not the ISA table order)
// A (16x32, MxK): lane&15 = M; lane<16 holds K = k0..k0+15 contiguous,
//                 lane>=16 holds K = k0+16..k0+31 contiguous
// B (32x16, KxN): lane&15 = N; lane<16 holds K = k0..k0+15 contiguous,
//                 lane>=16 holds K = k0+16..k0+31 contiguous
// ---------------------------------------------------------------------------
__device__ __forceinline__ f16x16 a_frag_from_f32(const float* __restrict__ rowPtr,
                                                  int k0, int half) {
  const float* p0 = rowPtr + k0 + half * 16;
  const float* p1 = rowPtr + k0 + half * 16 + 8;
  f32x4 f0 = *(const f32x4*)(p0);
  f32x4 f1 = *(const f32x4*)(p0 + 4);
  f32x4 f2 = *(const f32x4*)(p1);
  f32x4 f3 = *(const f32x4*)(p1 + 4);
  f16x16 a;
#pragma unroll
  for (int i = 0; i < 4; ++i) {
    a[i]      = (_Float16)f0[i];
    a[4 + i]  = (_Float16)f1[i];
    a[8 + i]  = (_Float16)f2[i];
    a[12 + i] = (_Float16)f3[i];
  }
  return a;
}

__device__ __forceinline__ f16x16 b_frag_from_f32(const float* __restrict__ p) {
  f16x16 b;
#pragma unroll
  for (int i = 0; i < 16; i += 4) {
    f32x4 f = *(const f32x4*)(p + i);
    b[i]     = (_Float16)f[0];
    b[i + 1] = (_Float16)f[1];
    b[i + 2] = (_Float16)f[2];
    b[i + 3] = (_Float16)f[3];
  }
  return b;
}

__device__ __forceinline__ f16x16 b_frag_from_f16(const _Float16* __restrict__ p) {
  f16x8 lo = *(const f16x8*)(p);
  f16x8 hi = *(const f16x8*)(p + 8);
  f16x16 b;
#pragma unroll
  for (int i = 0; i < 8; ++i) { b[i] = lo[i]; b[8 + i] = hi[i]; }
  return b;
}

// ---------------------------------------------------------------------------
// Kernel 0: one-time fp32 -> f16 conversion of W1 and W2 into workspace.
// ---------------------------------------------------------------------------
__global__ __launch_bounds__(256) void k0_convert_weights(
    const float* __restrict__ W1, const float* __restrict__ W2,
    _Float16* __restrict__ W1h, _Float16* __restrict__ W2h) {
  const int n = C_ * E_;                       // 524288 elements each
  const int j = (blockIdx.x * 256 + threadIdx.x) * 4;
  if (j < n) {
    f32x4 v = *(const f32x4*)(W1 + j);
    f16x4 h; h[0]=(_Float16)v[0]; h[1]=(_Float16)v[1]; h[2]=(_Float16)v[2]; h[3]=(_Float16)v[3];
    *(f16x4*)(W1h + j) = h;
  } else {
    const int jj = j - n;
    f32x4 v = *(const f32x4*)(W2 + jj);
    f16x4 h; h[0]=(_Float16)v[0]; h[1]=(_Float16)v[1]; h[2]=(_Float16)v[2]; h[3]=(_Float16)v[3];
    *(f16x4*)(W2h + jj) = h;
  }
}

// ---------------------------------------------------------------------------
// Kernel 1: logits = (x @ W1^T + b1)*3 + class_offsets    [16x1024 per block]
// Stored into the softmax output region (overwritten in-place by kernel 2).
// Two variants: f16 weights from workspace (fast) / fp32 weights (fallback).
// ---------------------------------------------------------------------------
template <typename WT>
__device__ __forceinline__ void gemm1_body(
    const float* __restrict__ x, const WT* __restrict__ W1,
    const float* __restrict__ b1, const float* __restrict__ coff,
    float* __restrict__ logits) {
  const int tid  = threadIdx.x;
  const int lane = tid & 31;
  const int wave = tid >> 5;
  const int half = lane >> 4;
  const int r    = lane & 15;
  const int m0   = blockIdx.x * 16;

  f32x8 acc[8] = {};
  const float* xrow = x + (size_t)(m0 + r) * E_;

#pragma unroll 1
  for (int kk = 0; kk < E_ / 32; ++kk) {
    const int k0 = kk * 32;
    f16x16 a = a_frag_from_f32(xrow, k0, half);
#pragma unroll
    for (int t = 0; t < 8; ++t) {
      const int c = wave * 128 + t * 16 + r;
      const WT* bp = W1 + (size_t)c * E_ + k0 + half * 16;
      f16x16 b;
      if constexpr (sizeof(WT) == 2) b = b_frag_from_f16((const _Float16*)bp);
      else                           b = b_frag_from_f32((const float*)bp);
      acc[t] = __builtin_amdgcn_wmma_f32_16x16x32_f16(
          false, a, false, b, (short)0, acc[t], false, false);
    }
  }
  // Epilogue: C/D layout -> row = rr + 8*half, col = tile + (lane&15)
#pragma unroll
  for (int t = 0; t < 8; ++t) {
    const int c  = wave * 128 + t * 16 + r;
    const float bb = b1[c];
    const float co = coff[c];
#pragma unroll
    for (int rr = 0; rr < 8; ++rr) {
      const int row = m0 + rr + 8 * half;
      logits[(size_t)row * C_ + c] = (acc[t][rr] + bb) * 3.0f + co;
    }
  }
}

__global__ __launch_bounds__(256) void k1_gemm_logits_f16w(
    const float* __restrict__ x, const _Float16* __restrict__ W1h,
    const float* __restrict__ b1, const float* __restrict__ coff,
    float* __restrict__ logits) {
  gemm1_body<_Float16>(x, W1h, b1, coff, logits);
}

__global__ __launch_bounds__(256) void k1_gemm_logits_f32w(
    const float* __restrict__ x, const float* __restrict__ W1,
    const float* __restrict__ b1, const float* __restrict__ coff,
    float* __restrict__ logits) {
  gemm1_body<float>(x, W1, b1, coff, logits);
}

// ---------------------------------------------------------------------------
// Kernel 2: per-(sn, group) softmax (in place over logits) + Gumbel sampling.
// argmax(log_softmax + g) == argmax(logits + g)  (lse is constant per group)
// sampled(forward) = soft + (hard - soft)
// ---------------------------------------------------------------------------
__global__ __launch_bounds__(256) void k2_softmax_sample(
    float* __restrict__ soft_io,                // in: logits, out: softmax
    const float* __restrict__ g1, const float* __restrict__ g2,
    const float* __restrict__ w_interp, const float* __restrict__ u_interp,
    float* __restrict__ sampled) {
  const int idx = blockIdx.x * blockDim.x + threadIdx.x;  // 0 .. M_*G_-1
  const int sn  = idx >> 5;
  const int g   = idx & 31;
  const size_t base = (size_t)sn * C_ + (size_t)g * CG_;

  float lg[32];
#pragma unroll
  for (int i = 0; i < 32; i += 4) {
    f32x4 v = *(const f32x4*)(soft_io + base + i);
    lg[i] = v[0]; lg[i + 1] = v[1]; lg[i + 2] = v[2]; lg[i + 3] = v[3];
  }
  float mx = lg[0];
#pragma unroll
  for (int i = 1; i < 32; ++i) mx = fmaxf(mx, lg[i]);
  float p[32];
  float sum = 0.0f;
#pragma unroll
  for (int i = 0; i < 32; ++i) { p[i] = __expf(lg[i] - mx); sum += p[i]; }
  const float inv = 1.0f / sum;

  int i1 = 0, i2 = 0;
  float bv1 = -3.0e38f, bv2 = -3.0e38f;
#pragma unroll
  for (int i = 0; i < 32; ++i) {   // strict '>' keeps first max (jnp.argmax)
    const float v1 = lg[i] + g1[base + i];
    const float v2 = lg[i] + g2[base + i];
    if (v1 > bv1) { bv1 = v1; i1 = i; }
    if (v2 > bv2) { bv2 = v2; i2 = i; }
  }
  const float w = w_interp[(size_t)sn * G_ + g];
  const float u = u_interp[(size_t)sn * G_ + g];
  const bool doInterp = (u < 1.0f);   // INTERP_PROB == 1.0
#pragma unroll
  for (int i = 0; i < 32; ++i) {
    const float soft = p[i] * inv;
    const float ohi  = (i == i1) ? 1.0f : 0.0f;
    const float ohj  = (i == i2) ? 1.0f : 0.0f;
    const float hard = doInterp ? (w * ohi + (1.0f - w) * ohj) : ohi;
    soft_io[base + i] = soft;
    sampled[base + i] = soft + (hard - soft);   // straight-through forward
  }
}

// ---------------------------------------------------------------------------
// Kernel 3: pos = LN(sampled @ W2^T + b2); neg == pos (reverse_grad forward
// is identity). Fused GEMM2 + LayerNorm; LDS reused f16 A-tile -> f32 H-tile.
// ---------------------------------------------------------------------------
template <typename WT>
__device__ __forceinline__ void gemm2_ln_body(
    const float* __restrict__ sampled, const WT* __restrict__ W2,
    const float* __restrict__ b2, const float* __restrict__ gamma,
    const float* __restrict__ beta,
    float* __restrict__ pos, float* __restrict__ neg) {
  __shared__ __align__(16) unsigned char smem[16 * C_ * 2];  // 32 KB
  _Float16* As = (_Float16*)smem;   // 16 x 1024 f16 (A operand)
  float*    Hs = (float*)smem;      // 16 x 512  f32 (reused after GEMM)
  __shared__ float redS[16][16];
  __shared__ float redQ[16][16];
  __shared__ float muA[16], rsA[16];

  const int tid  = threadIdx.x;
  const int lane = tid & 31;
  const int wave = tid >> 5;
  const int half = lane >> 4;
  const int r    = lane & 15;
  const int m0   = blockIdx.x * 16;

  // Stage sampled tile into LDS as f16 (coalesced)
  for (int i = tid; i < 16 * C_; i += 256) {
    const int row = i >> 10;
    const int c   = i & (C_ - 1);
    As[i] = (_Float16)sampled[(size_t)(m0 + row) * C_ + c];
  }
  __syncthreads();

  f32x8 acc[4] = {};
  const _Float16* arow = As + r * C_;
#pragma unroll 1
  for (int kk = 0; kk < C_ / 32; ++kk) {
    const int k0 = kk * 32;
    f16x8 x0 = *(const f16x8*)(arow + k0 + half * 16);
    f16x8 x1 = *(const f16x8*)(arow + k0 + half * 16 + 8);
    f16x16 a;
#pragma unroll
    for (int i = 0; i < 8; ++i) { a[i] = x0[i]; a[8 + i] = x1[i]; }
#pragma unroll
    for (int t = 0; t < 4; ++t) {
      const int e = wave * 64 + t * 16 + r;
      const WT* bp = W2 + (size_t)e * C_ + k0 + half * 16;
      f16x16 b;
      if constexpr (sizeof(WT) == 2) b = b_frag_from_f16((const _Float16*)bp);
      else                           b = b_frag_from_f32((const float*)bp);
      acc[t] = __builtin_amdgcn_wmma_f32_16x16x32_f16(
          false, a, false, b, (short)0, acc[t], false, false);
    }
  }
  __syncthreads();  // all reads of As done before reuse as Hs

  // h = acc + b2  ->  Hs[row][e]
#pragma unroll
  for (int t = 0; t < 4; ++t) {
    const int e  = wave * 64 + t * 16 + r;
    const float bb = b2[e];
#pragma unroll
    for (int rr = 0; rr < 8; ++rr) {
      const int row = rr + 8 * half;
      Hs[row * E_ + e] = acc[t][rr] + bb;
    }
  }
  __syncthreads();

  // LayerNorm: 16 threads per row, 32 elems per thread
  {
    const int row = tid >> 4;
    const int seg = tid & 15;
    float s1 = 0.0f, s2 = 0.0f;
    const float* hp = Hs + row * E_ + seg * 32;
#pragma unroll
    for (int i = 0; i < 32; ++i) { const float v = hp[i]; s1 += v; s2 += v * v; }
    redS[row][seg] = s1;
    redQ[row][seg] = s2;
  }
  __syncthreads();
  if (tid < 16) {
    float s1 = 0.0f, s2 = 0.0f;
#pragma unroll
    for (int i = 0; i < 16; ++i) { s1 += redS[tid][i]; s2 += redQ[tid][i]; }
    const float mu  = s1 * (1.0f / E_);
    const float var = s2 * (1.0f / E_) - mu * mu;
    muA[tid] = mu;
    rsA[tid] = rsqrtf(var + 1e-5f);
  }
  __syncthreads();
  for (int i = tid; i < 16 * E_; i += 256) {
    const int row = i >> 9;
    const int e   = i & (E_ - 1);
    const float o = (Hs[i] - muA[row]) * rsA[row] * gamma[e] + beta[e];
    const size_t gi = (size_t)(m0 + row) * E_ + e;
    pos[gi] = o;
    neg[gi] = o;   // reverse_gradient forward == identity
  }
}

__global__ __launch_bounds__(256) void k3_gemm2_ln_f16w(
    const float* __restrict__ sampled, const _Float16* __restrict__ W2h,
    const float* __restrict__ b2, const float* __restrict__ gamma,
    const float* __restrict__ beta,
    float* __restrict__ pos, float* __restrict__ neg) {
  gemm2_ln_body<_Float16>(sampled, W2h, b2, gamma, beta, pos, neg);
}

__global__ __launch_bounds__(256) void k3_gemm2_ln_f32w(
    const float* __restrict__ sampled, const float* __restrict__ W2,
    const float* __restrict__ b2, const float* __restrict__ gamma,
    const float* __restrict__ beta,
    float* __restrict__ pos, float* __restrict__ neg) {
  gemm2_ln_body<float>(sampled, W2, b2, gamma, beta, pos, neg);
}

// ---------------------------------------------------------------------------
extern "C" void kernel_launch(void* const* d_in, const int* in_sizes, int n_in,
                              void* d_out, int out_size, void* d_ws, size_t ws_size,
                              hipStream_t stream) {
  (void)in_sizes; (void)n_in; (void)out_size;

  const float* x     = (const float*)d_in[0];
  const float* W1    = (const float*)d_in[1];
  const float* b1    = (const float*)d_in[2];
  const float* coff  = (const float*)d_in[3];
  const float* W2    = (const float*)d_in[4];
  const float* b2    = (const float*)d_in[5];
  const float* gamma = (const float*)d_in[6];
  const float* beta  = (const float*)d_in[7];
  const float* g1    = (const float*)d_in[8];
  const float* g2    = (const float*)d_in[9];
  const float* wI    = (const float*)d_in[10];
  const float* uI    = (const float*)d_in[11];

  float* out        = (float*)d_out;
  float* outSampled = out;                                // (S,N,C)
  float* outSoft    = out + (size_t)M_ * C_;              // (S,N,C)
  float* outPos     = out + 2 * (size_t)M_ * C_;          // (S,N,E)
  float* outNeg     = outPos + (size_t)M_ * E_;           // (S,N,E)

  const size_t nW = (size_t)C_ * E_;                      // 524288 per matrix
  const bool useWs = (d_ws != nullptr) && (ws_size >= 2 * nW * sizeof(_Float16));

  if (useWs) {
    _Float16* W1h = (_Float16*)d_ws;
    _Float16* W2h = W1h + nW;
    // 0) one-time weight conversion fp32 -> f16 (2 MB workspace)
    k0_convert_weights<<<(int)((2 * nW / 4) / 256), 256, 0, stream>>>(W1, W2, W1h, W2h);
    // 1) GEMM1 -> logits parked in the softmax region
    k1_gemm_logits_f16w<<<M_ / 16, 256, 0, stream>>>(x, W1h, b1, coff, outSoft);
    // 2) softmax (in place) + Gumbel straight-through sampling
    k2_softmax_sample<<<(M_ * G_) / 256, 256, 0, stream>>>(outSoft, g1, g2, wI, uI,
                                                           outSampled);
    // 3) GEMM2 + LayerNorm fused; neg == pos
    k3_gemm2_ln_f16w<<<M_ / 16, 256, 0, stream>>>(outSampled, W2h, b2, gamma, beta,
                                                  outPos, outNeg);
  } else {
    k1_gemm_logits_f32w<<<M_ / 16, 256, 0, stream>>>(x, W1, b1, coff, outSoft);
    k2_softmax_sample<<<(M_ * G_) / 256, 256, 0, stream>>>(outSoft, g1, g2, wI, uI,
                                                           outSampled);
    k3_gemm2_ln_f32w<<<M_ / 16, 256, 0, stream>>>(outSampled, W2, b2, gamma, beta,
                                                  outPos, outNeg);
  }
}
